// FLUKEPlusModel_76836964925925
// MI455X (gfx1250) — matrix operand order repaired and checked by reading, not compiled, not run on hardware
//
#include <hip/hip_runtime.h>
#include <hip/hip_fp16.h>

typedef __attribute__((ext_vector_type(16))) _Float16 v16h;
typedef __attribute__((ext_vector_type(8)))  float    v8f;
typedef __attribute__((ext_vector_type(4)))  unsigned int v4u;

#define NQS 32
#define NDT 180
#define NDP 192      // padded token count (12 WMMA tiles of 16)
#define DD  128
#define HSTR 136     // f16 row stride in halves (272B = 68 dwords -> bank spread)
#define SSTR 193     // S row stride in floats (odd -> conflict-free per-q scans)
#define NEGV -1000000000.0f

union Frag { v4u u[2]; v16h h; };

struct SMem {
  _Float16 Qh[NQS*HSTR];        // 8.5 KB  queries f16 (padded rows)
  _Float16 Dh[NDP*HSTR];        // 51 KB   doc tokens f16 (rows 180..191 zeroed)
  float    S[NQS*SSTR];         // 24.7 KB similarity matrix
  float    nn[NDP], g1[NDP], gg[NDP];   // token self/adjacent dots
  float    rs2[NDP], rs3[NDP];          // window rsqrt normalizers
  float    dmf[NDP];                    // doc mask as float
  float    scr[NQS][8][5];              // per-q partial top3/max2/max3
  float    ptsA[NQS], pmaxA[NQS], m2A[NQS], m3A[NQS], padA[NQS];
};

__global__ __launch_bounds__(256)
void fluke_score_kernel(const float* __restrict__ qe,
                        const float* __restrict__ de,
                        const float* __restrict__ iw,
                        const unsigned char* __restrict__ qmask,
                        const unsigned char* __restrict__ dmask,
                        const float* __restrict__ aw1, const float* __restrict__ ab1,
                        const float* __restrict__ aw2, const float* __restrict__ ab2,
                        const float* __restrict__ ablend,
                        const float* __restrict__ mgs,
                        const float* __restrict__ tw1, const float* __restrict__ tb1,
                        const float* __restrict__ tw2, const float* __restrict__ tb2,
                        float* __restrict__ out)
{
  __shared__ SMem sh;
  const int tid  = threadIdx.x;
  const int n    = blockIdx.x;
  const int lane = tid & 31;
  const int wave = tid >> 5;

  // Prefetch tail-phase weights (emits global_prefetch_b8; 8KB of tir_w1)
  if (tid < 128) __builtin_prefetch(tw1 + tid * 16, 0, 1);

  // ---- Phase A: load + f32->f16 convert into LDS (padded row stride) ----
  {
    const float2* q2 = (const float2*)qe;
    __half2* dst = (__half2*)sh.Qh;               // row stride HSTR/2 = 68
    for (int p = tid; p < NQS*DD/2; p += 256) {
      const int t = p >> 6, j = p & 63;           // 64 half2 per row
      float2 f = q2[p];
      dst[t*(HSTR/2) + j] = __floats2half2_rn(f.x, f.y);
    }
  }
  {
    const float2* d2 = (const float2*)(de + (size_t)n * NDT * DD);
    __half2* dst = (__half2*)sh.Dh;
    for (int p = tid; p < NDT*DD/2; p += 256) {
      const int t = p >> 6, j = p & 63;
      float2 f = d2[p];
      dst[t*(HSTR/2) + j] = __floats2half2_rn(f.x, f.y);
    }
    // zero tokens 180..191 (first 128 halves of each row; pad halves never read)
    for (int p = tid; p < (NDP-NDT)*DD/2; p += 256) {
      const int t = NDT + (p >> 6), j = p & 63;
      dst[t*(HSTR/2) + j] = __floats2half2_rn(0.f, 0.f);
    }
  }
  for (int t = tid; t < NDP; t += 256)
    sh.dmf[t] = (t < NDT && dmask[(size_t)n*NDT + t]) ? 1.0f : 0.0f;

  __syncthreads();

  // ---- Phase B1: token band-Gram (nn, d_t.d_{t+1}, d_t.d_{t+2}) on VALU ----
  if (tid < NDT) {
    const __half2* D2 = (const __half2*)sh.Dh;
    const __half2* a2 = D2 + (tid    )*(HSTR/2);
    const __half2* b2 = D2 + (tid + 1)*(HSTR/2);   // rows >=180 zeroed
    const __half2* c2 = D2 + (tid + 2)*(HSTR/2);
    float snn = 0.f, sg1 = 0.f, sgg = 0.f;
    for (int j = 0; j < DD/2; ++j) {
      float2 a = __half22float2(a2[j]);
      float2 b = __half22float2(b2[j]);
      float2 c = __half22float2(c2[j]);
      snn += a.x*a.x + a.y*a.y;
      sg1 += a.x*b.x + a.y*b.y;
      sgg += a.x*c.x + a.y*c.y;
    }
    sh.nn[tid] = snn; sh.g1[tid] = sg1; sh.gg[tid] = sgg;
  }

  // ---- Phase B2: S = Q x D^T via v_wmma_f32_16x16x32_f16 ----
  {
    const int qtile = (wave*3) / 12;     // waves 0-3 -> q rows 0..15, waves 4-7 -> 16..31
    const int hi    = lane >> 4;
    const int row   = qtile*16 + (lane & 15);
    // A fragments (ISA 16-bit A 16x32 layout): two aligned 16B chunks per k-step
    v16h afrag[4];
    #pragma unroll
    for (int ks = 0; ks < 4; ++ks) {
      const _Float16* ap = sh.Qh + row*HSTR + ks*32 + hi*8;
      Frag fa;
      fa.u[0] = *(const v4u*)(ap);        // K = kb + hi*8 .. +8
      fa.u[1] = *(const v4u*)(ap + 16);   // K = kb + 16 + hi*8 .. +8
      afrag[ks] = fa.h;
    }
    #pragma unroll
    for (int i = 0; i < 3; ++i) {
      const int tt   = (wave*3 + i) % 12;
      const int tcol = tt*16 + (lane & 15);
      v8f acc = {};
      #pragma unroll
      for (int ks = 0; ks < 4; ++ks) {
        // B (same K order as A): two aligned 16B chunks, K = kb + hi*8 and kb + 16 + hi*8
        const _Float16* bp = sh.Dh + tcol*HSTR + ks*32 + hi*8;
        Frag fb;
        fb.u[0] = *(const v4u*)(bp);
        fb.u[1] = *(const v4u*)(bp + 16);
        acc = __builtin_amdgcn_wmma_f32_16x16x32_f16(
                  false, afrag[ks], false, fb.h, (short)0, acc, false, false);
      }
      #pragma unroll
      for (int r = 0; r < 8; ++r)
        sh.S[(qtile*16 + hi*8 + r)*SSTR + tt*16 + (lane & 15)] = acc[r];
    }
  }
  __syncthreads();

  // ---- Phase C: window normalizers  q.l2norm(mean) = (sum S)*rsqrt(||sum d||^2 + k^2*1e-12) ----
  if (tid < NDT-1)
    sh.rs2[tid] = rsqrtf(sh.nn[tid] + sh.nn[tid+1] + 2.f*sh.g1[tid] + 4e-12f);
  if (tid < NDT-2)
    sh.rs3[tid] = rsqrtf(sh.nn[tid] + sh.nn[tid+1] + sh.nn[tid+2]
                         + 2.f*(sh.g1[tid] + sh.g1[tid+1] + sh.gg[tid]) + 9e-12f);
  __syncthreads();

  // ---- Phase D: per-(q, 1/8 slice) scans: top3, window maxima ----
  {
    const int q = tid >> 3, sub = tid & 7;
    const float* Sq = sh.S + q*SSTR;
    float t0 = NEGV, t1 = NEGV, t2 = NEGV;
    for (int t = sub; t < NDT; t += 8) {
      float v = sh.dmf[t] > 0.5f ? Sq[t] : NEGV;
      if (v > t0)      { t2 = t1; t1 = t0; t0 = v; }
      else if (v > t1) { t2 = t1; t1 = v; }
      else if (v > t2) { t2 = v; }
    }
    float mx2 = NEGV;
    for (int w = sub; w < NDT-1; w += 8) {
      float ok = sh.dmf[w]*sh.dmf[w+1];
      float v  = ok > 0.5f ? (Sq[w] + Sq[w+1]) * sh.rs2[w] : NEGV;
      mx2 = fmaxf(mx2, v);
    }
    float mx3 = NEGV;
    for (int w = sub; w < NDT-2; w += 8) {
      float ok = sh.dmf[w]*sh.dmf[w+1]*sh.dmf[w+2];
      float v  = ok > 0.5f ? (Sq[w] + Sq[w+1] + Sq[w+2]) * sh.rs3[w] : NEGV;
      mx3 = fmaxf(mx3, v);
    }
    sh.scr[q][sub][0] = t0; sh.scr[q][sub][1] = t1; sh.scr[q][sub][2] = t2;
    sh.scr[q][sub][3] = mx2; sh.scr[q][sub][4] = mx3;
  }
  __syncthreads();

  // ---- Phase E: merge per q, softmax(top3/0.1) -> pts ----
  if ((tid & 7) == 0) {
    const int q = tid >> 3;
    float t0 = NEGV, t1 = NEGV, t2 = NEGV, mx2 = NEGV, mx3 = NEGV;
    for (int s = 0; s < 8; ++s) {
      #pragma unroll
      for (int j = 0; j < 3; ++j) {
        float v = sh.scr[q][s][j];
        if (v > t0)      { t2 = t1; t1 = t0; t0 = v; }
        else if (v > t1) { t2 = t1; t1 = v; }
        else if (v > t2) { t2 = v; }
      }
      mx2 = fmaxf(mx2, sh.scr[q][s][3]);
      mx3 = fmaxf(mx3, sh.scr[q][s][4]);
    }
    float e1 = expf((t1 - t0)*10.0f), e2 = expf((t2 - t0)*10.0f);  // 1/TEMP = 10
    float pts = (t0 + t1*e1 + t2*e2) / (1.0f + e1 + e2);
    float qmf = qmask[q] ? 1.0f : 0.0f;
    sh.ptsA[q] = pts; sh.pmaxA[q] = t0; sh.m2A[q] = mx2; sh.m3A[q] = mx3;
    sh.padA[q] = pts * qmf;
  }
  __syncthreads();

  // ---- Phase F: wave-0 reductions + MLPs + final blend ----
  if (tid < 32) {
    const int q = tid;
    float qmf  = qmask[q] ? 1.0f : 0.0f;
    float wqv  = iw[q] * qmf;
    float pts  = sh.ptsA[q], pmax = sh.pmaxA[q];
    float base = wqv*pts, m1 = wqv*pmax, m2 = wqv*sh.m2A[q], m3 = wqv*sh.m3A[q];
    float nv = qmf, sp = pmax*qmf, sq = pmax*pmax*qmf;
    float mxv = qmf > 0.5f ? pmax : NEGV;
    #pragma unroll
    for (int m = 16; m >= 1; m >>= 1) {
      base += __shfl_xor(base, m, 32);
      m1   += __shfl_xor(m1,   m, 32);
      m2   += __shfl_xor(m2,   m, 32);
      m3   += __shfl_xor(m3,   m, 32);
      nv   += __shfl_xor(nv,   m, 32);
      sp   += __shfl_xor(sp,   m, 32);
      sq   += __shfl_xor(sq,   m, 32);
      mxv   = fmaxf(mxv, __shfl_xor(mxv, m, 32));
    }
    float mean = sp / nv;
    float stdv = sqrtf(fmaxf(sq/nv - mean*mean, 0.0f) + 1e-6f);
    float frac = nv * (1.0f/32.0f);
    // ASC MLP 4->32->1 (lane = hidden unit)
    float h = mean*aw1[q] + mxv*aw1[32+q] + stdv*aw1[64+q] + frac*aw1[96+q] + ab1[q];
    h = fmaxf(h, 0.0f);
    float cs = h * aw2[q];
    #pragma unroll
    for (int m = 16; m >= 1; m >>= 1) cs += __shfl_xor(cs, m, 32);
    float calib     = tanhf(cs + ab2[0]);
    float asc_score = m1 * (1.0f + calib);
    float blend     = 1.0f / (1.0f + expf(-ablend[0]));
    // MGS gate softmax (3)
    float l0 = mgs[0], l1 = mgs[1], l2 = mgs[2];
    float lm = fmaxf(l0, fmaxf(l1, l2));
    float e0 = expf(l0-lm), e1 = expf(l1-lm), e2 = expf(l2-lm);
    float es = e0 + e1 + e2;
    // TIR MLP 32->64->1 (lane handles hidden j and j+32)
    float hA = tb1[q], hB = tb1[q+32];
    for (int qq = 0; qq < 32; ++qq) {
      float p = sh.padA[qq];
      hA += p * tw1[qq*64 + q];
      hB += p * tw1[qq*64 + q + 32];
    }
    hA = fmaxf(hA, 0.0f); hB = fmaxf(hB, 0.0f);
    float tv = hA*tw2[q] + hB*tw2[q+32];
    #pragma unroll
    for (int m = 16; m >= 1; m >>= 1) tv += __shfl_xor(tv, m, 32);
    tv += tb2[0];

    float total = blend*base + (1.0f - blend)*asc_score
                + (e0/es)*m1 + (e1/es)*m2 + (e2/es)*m3 + tv;
    if (q == 0) out[n] = total;
  }
}

extern "C" void kernel_launch(void* const* d_in, const int* in_sizes, int n_in,
                              void* d_out, int out_size, void* d_ws, size_t ws_size,
                              hipStream_t stream) {
  (void)in_sizes; (void)n_in; (void)out_size; (void)d_ws; (void)ws_size;
  const float* qe  = (const float*)d_in[0];
  const float* de  = (const float*)d_in[1];
  const float* iw  = (const float*)d_in[2];
  const unsigned char* qm = (const unsigned char*)d_in[3];
  const unsigned char* dm = (const unsigned char*)d_in[4];
  const float* aw1 = (const float*)d_in[5];
  const float* ab1 = (const float*)d_in[6];
  const float* aw2 = (const float*)d_in[7];
  const float* ab2 = (const float*)d_in[8];
  const float* abl = (const float*)d_in[9];
  const float* mgs = (const float*)d_in[10];
  const float* tw1 = (const float*)d_in[11];
  const float* tb1 = (const float*)d_in[12];
  const float* tw2 = (const float*)d_in[13];
  const float* tb2 = (const float*)d_in[14];
  float* out = (float*)d_out;

  hipLaunchKernelGGL(fluke_score_kernel, dim3(4096), dim3(256), 0, stream,
                     qe, de, iw, qm, dm, aw1, ab1, aw2, ab2, abl, mgs,
                     tw1, tb1, tw2, tb2, out);
}
